// Attention_37538014167301
// MI455X (gfx1250) — matrix operand order repaired and checked by reading, not compiled, not run on hardware
//
#include <hip/hip_runtime.h>
#include <hip/hip_bf16.h>

#define B_   2
#define S_   2048
#define E_   2048
#define H_   16
#define KVH_ 4
#define D_   128
#define NREP_ (H_ / KVH_)

typedef __attribute__((ext_vector_type(16))) __bf16 v16bf;
typedef __attribute__((ext_vector_type(8)))  __bf16 v8bf;
typedef __attribute__((ext_vector_type(4)))  __bf16 v4bf;
typedef __attribute__((ext_vector_type(8)))  float  v8f;
typedef __attribute__((ext_vector_type(4)))  unsigned int v4u;
typedef __attribute__((ext_vector_type(8)))  int v8i;
typedef __attribute__((ext_vector_type(4)))  int v4i;

// A-operand fragment (16-bit, 16xK): lane<16 takes K {0..15}, lane>=16 takes {16..31}
__device__ __forceinline__ v16bf load_a_frag(const __bf16* p, int hi) {
    return *(const v16bf*)(p + hi * 16);
}

// ---------------------------------------------------------------------------
// TDM: 2-D tile load Global -> LDS via Tensor Data Mover (D# per ISA 8.3/8.4).
// tile_w elements per row (2-byte), tile_h rows, row stride in elements.
// 6-arg builtin variant: (v4u, v8i, v4i, v4i, v8i, i32 cpol).
// ---------------------------------------------------------------------------
__device__ __forceinline__ void tdm_load_2d_bf16(
    const __bf16* gptr, unsigned lds_off, int tile_w, int tile_h, int row_stride)
{
    unsigned long long ga = (unsigned long long)gptr;
    v4u g0;
    g0[0] = 1u;                                           // count=1, user descriptor
    g0[1] = lds_off;                                      // lds_addr (bytes)
    g0[2] = (unsigned)(ga & 0xFFFFFFFFu);                 // global_addr[31:0]
    g0[3] = (unsigned)((ga >> 32) & 0x1FFFFFFu) | (2u << 30);  // addr[56:32] | type=2
    v8i g1;
    g1[0] = (int)(1u << 16);                              // data_size=1 (2 bytes)
    g1[1] = (int)(((unsigned)row_stride & 0xFFFFu) << 16);            // tensor_dim0[15:0]
    g1[2] = (int)((((unsigned)row_stride >> 16) & 0xFFFFu) |
                  (((unsigned)tile_h & 0xFFFFu) << 16));              // dim0[31:16] | tensor_dim1[15:0]
    g1[3] = (int)(((unsigned)tile_w & 0xFFFFu) << 16);                // tensor_dim1[31:16]=0 | tile_dim0
    g1[4] = (int)((unsigned)tile_h & 0xFFFFu);                        // tile_dim1 | tile_dim2=0
    g1[5] = (int)(unsigned)row_stride;                                // tensor_dim0_stride[31:0]
    g1[6] = 0;                                                        // stride[47:32] | dim1_stride lo
    g1[7] = 0;
    v4i z4 = {};
    v8i z8 = {};
    __builtin_amdgcn_tensor_load_to_lds(g0, g1, z4, z4, z8, 0);
}

// ---------------------------------------------------------------------------
// Elementwise f32 -> bf16 (hoists ALL conversion cost out of GEMM hot loops).
// ---------------------------------------------------------------------------
__global__ void __launch_bounds__(256) cvt_f32_bf16(
    const float* __restrict__ in, __bf16* __restrict__ out, int n4)
{
    int t = blockIdx.x * blockDim.x + threadIdx.x;
    if (t >= n4) return;
    float4 v = *(const float4*)(in + (size_t)t * 4);
    v4bf o;
    o[0] = (__bf16)v.x; o[1] = (__bf16)v.y; o[2] = (__bf16)v.z; o[3] = (__bf16)v.w;
    *(v4bf*)(out + (size_t)t * 4) = o;
}

// ---------------------------------------------------------------------------
// GEMM: C[M,N] = A[M,K] (bf16) x W[N,K]^T (bf16), bf16 WMMA, f32 accumulate.
// Block tile 128x64 (4 waves x 32x64). Shared 64x32 W tile staged in LDS by
// the Tensor Data Mover, double-buffered (TENSORcnt + barriers).
// ---------------------------------------------------------------------------
template <typename OutT>
__global__ void __launch_bounds__(128) gemm_bf16_wmma(
    const __bf16* __restrict__ A, const __bf16* __restrict__ W,
    OutT* __restrict__ C, int M, int N, int K)
{
    __shared__ __bf16 smem[2][64 * 32];   // two 4KB W-tile buffers

    const int tid  = threadIdx.x;
    const int lane = tid & 31;
    const int widx = tid >> 5;
    const int nt64 = N >> 6;
    const int mtb  = blockIdx.x / nt64;   // 128-row block tile
    const int ntb  = blockIdx.x % nt64;   // 64-col block tile
    const int l16  = lane & 15;
    const int hi   = lane >> 4;

    const int m0 = mtb * 128 + widx * 32;
    const __bf16* arow0 = A + (size_t)(m0 + l16) * K;
    const __bf16* arow1 = A + (size_t)(m0 + 16 + l16) * K;
    const __bf16* wbase = W + (size_t)(ntb * 64) * K;

    const unsigned lds0 = (unsigned)(unsigned long long)&smem[0][0];
    const unsigned lds1 = (unsigned)(unsigned long long)&smem[1][0];

    if (tid < 32) tdm_load_2d_bf16(wbase, lds0, 32, 64, K);   // prologue: k=0 -> buf0

    v8f acc[8] = {};

    for (int i = 0, k = 0; k < K; k += 32, ++i) {
        if (tid < 32) {
            if (k + 32 < K) {
                tdm_load_2d_bf16(wbase + k + 32, (i & 1) ? lds0 : lds1, 32, 64, K);
                __builtin_amdgcn_s_wait_tensorcnt(1);   // current buffer's TDM done
            } else {
                __builtin_amdgcn_s_wait_tensorcnt(0);
            }
        }
        __syncthreads();                                // publish buf[i&1]

        const __bf16* bt_ = &smem[i & 1][0];
        v16bf af0 = load_a_frag(arow0 + k, hi);
        v16bf af1 = load_a_frag(arow1 + k, hi);
        if (k + 256 < K) __builtin_prefetch(arow0 + k + 256, 0, 0);
#pragma unroll
        for (int sub = 0; sub < 4; ++sub) {
            // B fragment: column n = W row; K 0..15 (lane<16) / 16..31 (lane>=16)
            v16bf bfrag = *(const v16bf*)(bt_ + (sub * 16 + l16) * 32 + hi * 16);
            acc[sub] = __builtin_amdgcn_wmma_f32_16x16x32_bf16(
                false, af0, false, bfrag, (short)0, acc[sub], false, false);
            acc[4 + sub] = __builtin_amdgcn_wmma_f32_16x16x32_bf16(
                false, af1, false, bfrag, (short)0, acc[4 + sub], false, false);
        }
        __syncthreads();                                // all reads done before overwrite
    }

    // C layout: lane = column, vgpr r = row (r + 8*hi)
#pragma unroll
    for (int half = 0; half < 2; ++half) {
        OutT* crow = C + (size_t)(m0 + half * 16 + hi * 8) * N + ntb * 64 + l16;
#pragma unroll
        for (int sub = 0; sub < 4; ++sub)
#pragma unroll
            for (int r = 0; r < 8; ++r)
                crow[(size_t)r * N + sub * 16] = (OutT)acc[half * 4 + sub][r];
    }
}

// ---------------------------------------------------------------------------
// RoPE + transpose: [b][s][h][d] bf16 -> [b][h][s][d] bf16
// ---------------------------------------------------------------------------
__global__ void __launch_bounds__(256) rope_bf16_T(
    const __bf16* __restrict__ in, __bf16* __restrict__ out,
    const float* __restrict__ cosb, const float* __restrict__ sinb,
    int nheads, int apply_rope)
{
    const int half_d = D_ / 2;
    int t = blockIdx.x * blockDim.x + threadIdx.x;
    int total = B_ * S_ * nheads * half_d;
    if (t >= total) return;
    int dp = t % half_d;
    int h  = (t / half_d) % nheads;
    int s  = (t / (half_d * nheads)) % S_;
    int b  =  t / (half_d * nheads * S_);
    const __bf16* src = in + ((size_t)(b * S_ + s) * nheads + h) * D_ + 2 * dp;
    float xr = (float)src[0], xi = (float)src[1];
    float orr = xr, oi = xi;
    if (apply_rope) {
        float c  = cosb[s * half_d + dp];
        float sn = sinb[s * half_d + dp];
        orr = xr * c - xi * sn;
        oi  = xr * sn + xi * c;
    }
    __bf16* dst = out + ((size_t)(b * nheads + h) * S_ + s) * D_ + 2 * dp;
    dst[0] = (__bf16)orr;
    dst[1] = (__bf16)oi;
}

// V: [b][s][kvh][d] bf16 -> [b][kvh][d][s] bf16 (V^T rows contiguous in s)
__global__ void __launch_bounds__(256) v_transpose_bf16(
    const __bf16* __restrict__ in, __bf16* __restrict__ out)
{
    int t = blockIdx.x * blockDim.x + threadIdx.x;
    int total = B_ * S_ * KVH_ * D_;
    if (t >= total) return;
    int d = t % D_;
    int h = (t / D_) % KVH_;
    int s = (t / (D_ * KVH_)) % S_;
    int b =  t / (D_ * KVH_ * S_);
    out[((size_t)(b * KVH_ + h) * D_ + d) * S_ + s] =
        in[((size_t)(b * S_ + s) * KVH_ + h) * D_ + d];
}

// ---------------------------------------------------------------------------
// Flash attention: one wave per (b, h, 16-query tile).
// S^T = K x Q^T (lane = query -> lane-local softmax stats),
// out^T = V^T x P^T (P^T fragment built with 16 half-wave shuffles).
// ---------------------------------------------------------------------------
__global__ void __launch_bounds__(128) flash_attn_wmma(
    const __bf16* __restrict__ qT, const __bf16* __restrict__ kT,
    const __bf16* __restrict__ vT, __bf16* __restrict__ outp)
{
    const int lane = threadIdx.x & 31;
    const int wave = blockIdx.x * (blockDim.x >> 5) + (threadIdx.x >> 5);
    const int QT = S_ / 16;
    const int qt = wave % QT;
    const int h  = (wave / QT) % H_;
    const int b  =  wave / (QT * H_);
    const int hkv = h % KVH_;              // matches reference jnp.tile
    const int l16 = lane & 15;
    const int hi  = lane >> 4;
    const int qbase = qt * 16;
    const float scale = 0.08838834764831845f;  // 1/sqrt(128)

    const __bf16* qrow = qT + ((size_t)(b * H_ + h) * S_ + qbase + l16) * D_;
    v16bf qf[4];
#pragma unroll
    for (int c = 0; c < 4; ++c)
        qf[c] = *(const v16bf*)(qrow + c * 32 + hi * 16);

    const __bf16* kbp = kT + (size_t)(b * KVH_ + hkv) * S_ * D_;
    const __bf16* vbp = vT + (size_t)(b * KVH_ + hkv) * D_ * S_;

    v8f acc[8] = {};
    float m = -__builtin_inff();
    float l = 0.0f;
    const int qg = qbase + l16;

    for (int kb = 0; kb < qbase + 16; kb += 32) {
        v8f s0 = {}, s1 = {};
        {
            const __bf16* krow = kbp + (size_t)(kb + l16) * D_;
#pragma unroll
            for (int c = 0; c < 4; ++c)
                s0 = __builtin_amdgcn_wmma_f32_16x16x32_bf16(
                    false, load_a_frag(krow + c * 32, hi), false, qf[c],
                    (short)0, s0, false, false);
        }
        {
            const __bf16* krow = kbp + (size_t)(kb + 16 + l16) * D_;
#pragma unroll
            for (int c = 0; c < 4; ++c)
                s1 = __builtin_amdgcn_wmma_f32_16x16x32_bf16(
                    false, load_a_frag(krow + c * 32, hi), false, qf[c],
                    (short)0, s1, false, false);
        }
        float ninf = -__builtin_inff();
#pragma unroll
        for (int r = 0; r < 8; ++r) {
            int k0 = kb + r + 8 * hi;
            s0[r] = (k0 <= qg)      ? s0[r] * scale : ninf;
            s1[r] = (k0 + 16 <= qg) ? s1[r] * scale : ninf;
        }
        float mloc = s0[0];
#pragma unroll
        for (int r = 0; r < 8; ++r) { mloc = fmaxf(mloc, s0[r]); mloc = fmaxf(mloc, s1[r]); }
        mloc = fmaxf(mloc, __shfl_xor(mloc, 16, 32));
        float mnew  = fmaxf(m, mloc);
        float alpha = __expf(m - mnew);
        float p0[8], p1[8], lsum = 0.0f;
#pragma unroll
        for (int r = 0; r < 8; ++r) {
            p0[r] = __expf(s0[r] - mnew);
            p1[r] = __expf(s1[r] - mnew);
            lsum += p0[r] + p1[r];
        }
        lsum += __shfl_xor(lsum, 16, 32);
        l = l * alpha + lsum;
        m = mnew;
#pragma unroll
        for (int t = 0; t < 8; ++t)
#pragma unroll
            for (int r = 0; r < 8; ++r) acc[t][r] *= alpha;

        float o0[8], o1[8];
#pragma unroll
        for (int r = 0; r < 8; ++r) {
            o0[r] = __shfl_xor(p0[r], 16, 32);
            o1[r] = __shfl_xor(p1[r], 16, 32);
        }
        v16bf pf;
#pragma unroll
        for (int r = 0; r < 8; ++r) {
            pf[r]     = (__bf16)(hi == 0 ? p0[r] : o1[r]);
            pf[r + 8] = (__bf16)(hi == 0 ? o0[r] : p1[r]);
        }
#pragma unroll
        for (int t = 0; t < 8; ++t) {
            const __bf16* vrow = vbp + (size_t)(t * 16 + l16) * S_ + kb;
            acc[t] = __builtin_amdgcn_wmma_f32_16x16x32_bf16(
                false, load_a_frag(vrow, hi), false, pf, (short)0, acc[t], false, false);
        }
    }

    float inv_l = 1.0f / l;
    __bf16* orow = outp + (size_t)(b * S_ + qbase + l16) * E_ + h * D_ + hi * 8;
#pragma unroll
    for (int t = 0; t < 8; ++t) {
        v8bf o;
#pragma unroll
        for (int r = 0; r < 8; ++r) o[r] = (__bf16)(acc[t][r] * inv_l);
        *(v8bf*)(orow + t * 16) = o;
    }
}

// ---------------------------------------------------------------------------
extern "C" void kernel_launch(void* const* d_in, const int* in_sizes, int n_in,
                              void* d_out, int out_size, void* d_ws, size_t ws_size,
                              hipStream_t stream)
{
    (void)in_sizes; (void)n_in; (void)out_size; (void)ws_size;
    const float* x    = (const float*)d_in[0];
    const float* wq   = (const float*)d_in[1];
    const float* wk   = (const float*)d_in[2];
    const float* wv   = (const float*)d_in[3];
    const float* wo   = (const float*)d_in[4];
    const float* fcos = (const float*)d_in[5];
    const float* fsin = (const float*)d_in[6];

    char* ws = (char*)d_ws;
    size_t off = 0;
    __bf16* x_bf  = (__bf16*)(ws + off); off += (size_t)B_ * S_ * E_      * 2;  // 16 MB
    __bf16* wq_bf = (__bf16*)(ws + off); off += (size_t)E_ * E_           * 2;  //  8 MB
    __bf16* wk_bf = (__bf16*)(ws + off); off += (size_t)KVH_ * D_ * E_    * 2;  //  2 MB
    __bf16* wv_bf = (__bf16*)(ws + off); off += (size_t)KVH_ * D_ * E_    * 2;  //  2 MB
    __bf16* wo_bf = (__bf16*)(ws + off); off += (size_t)E_ * E_           * 2;  //  8 MB
    __bf16* q_pb  = (__bf16*)(ws + off); off += (size_t)B_ * S_ * H_ * D_ * 2;  // 16 MB
    __bf16* k_pb  = (__bf16*)(ws + off); off += (size_t)B_ * S_ * KVH_ * D_ * 2;//  4 MB
    __bf16* v_pb  = (__bf16*)(ws + off); off += (size_t)B_ * S_ * KVH_ * D_ * 2;//  4 MB
    __bf16* qTb   = (__bf16*)(ws + off); off += (size_t)B_ * H_ * S_ * D_ * 2;  // 16 MB
    __bf16* kTb   = (__bf16*)(ws + off); off += (size_t)B_ * KVH_ * S_ * D_ * 2;//  4 MB
    __bf16* vTb   = (__bf16*)(ws + off); off += (size_t)B_ * KVH_ * D_ * S_ * 2;//  4 MB
    __bf16* attn_bf = x_bf;  // reuse x_bf region (x consumed by projections)

    const int M = B_ * S_;

    // Precision pre-pass: everything WMMA touches becomes bf16 once.
    cvt_f32_bf16<<<(B_ * S_ * E_ / 4) / 256,   256, 0, stream>>>(x,  x_bf,  B_ * S_ * E_ / 4);
    cvt_f32_bf16<<<(E_ * E_ / 4) / 256,        256, 0, stream>>>(wq, wq_bf, E_ * E_ / 4);
    cvt_f32_bf16<<<(KVH_ * D_ * E_ / 4) / 256, 256, 0, stream>>>(wk, wk_bf, KVH_ * D_ * E_ / 4);
    cvt_f32_bf16<<<(KVH_ * D_ * E_ / 4) / 256, 256, 0, stream>>>(wv, wv_bf, KVH_ * D_ * E_ / 4);
    cvt_f32_bf16<<<(E_ * E_ / 4) / 256,        256, 0, stream>>>(wo, wo_bf, E_ * E_ / 4);

    // Q/K/V projections (bf16 in, bf16 out), block tile 128x64
    gemm_bf16_wmma<__bf16><<<(M / 128) * ((H_   * D_) / 64), 128, 0, stream>>>(x_bf, wq_bf, q_pb, M, H_   * D_, E_);
    gemm_bf16_wmma<__bf16><<<(M / 128) * ((KVH_ * D_) / 64), 128, 0, stream>>>(x_bf, wk_bf, k_pb, M, KVH_ * D_, E_);
    gemm_bf16_wmma<__bf16><<<(M / 128) * ((KVH_ * D_) / 64), 128, 0, stream>>>(x_bf, wv_bf, v_pb, M, KVH_ * D_, E_);

    // RoPE + layout transforms
    rope_bf16_T<<<(B_ * S_ * H_   * (D_ / 2)) / 256, 256, 0, stream>>>(q_pb, qTb, fcos, fsin, H_,   1);
    rope_bf16_T<<<(B_ * S_ * KVH_ * (D_ / 2)) / 256, 256, 0, stream>>>(k_pb, kTb, fcos, fsin, KVH_, 1);
    v_transpose_bf16<<<(B_ * S_ * KVH_ * D_) / 256, 256, 0, stream>>>(v_pb, vTb);

    // Attention (writes bf16, aliases x_bf region)
    flash_attn_wmma<<<(B_ * H_ * (S_ / 16)) / 4, 128, 0, stream>>>(qTb, kTb, vTb, attn_bf);

    // Output projection (f32 out to d_out)
    gemm_bf16_wmma<float><<<(M / 128) * (E_ / 64), 128, 0, stream>>>(attn_bf, wo_bf, (float*)d_out, M, E_, E_);
}
